// MultiHeadAttention1_2250562863496
// MI455X (gfx1250) — compile-verified, matrix operand order checked by reading, not run on hardware
//
#include <hip/hip_runtime.h>
#include <math.h>

#define F_DIM 1024
#define T_DIM 2048
#define B_DIM 4
#define H_DIM 16
#define DK_DIM 64

typedef __bf16 bf16;
typedef __attribute__((ext_vector_type(16))) __bf16 bf16x16;
typedef __attribute__((ext_vector_type(8)))  float  floatx8;
typedef __attribute__((ext_vector_type(4)))  unsigned int uintx4;

union FragU { bf16x16 v; uintx4 q[2]; };

__device__ __forceinline__ floatx8 zero8() {
  floatx8 z = {0.f, 0.f, 0.f, 0.f, 0.f, 0.f, 0.f, 0.f};
  return z;
}

// Load a 16x32 bf16 WMMA fragment (A-layout; B uses the same per-lane pattern
// with "row" meaning the B column, i.e. pass the matrix holding B^T rows).
// Lane L: row = base_row + (L%16); 8 contiguous elems at col + (L/16)*8 and
// 8 more at col + 16 + (L/16)*8  -> two 16-byte global loads.
__device__ __forceinline__ bf16x16 load_frag(const bf16* base, int ld, int row, int col, int lane) {
  const bf16* p = base + (size_t)(row + (lane & 15)) * ld + col + ((lane >> 4) << 3);
  FragU f;
  f.q[0] = *(const uintx4*)p;
  f.q[1] = *(const uintx4*)(p + 16);
  return f.v;
}

__device__ __forceinline__ floatx8 wmma_bf16(bf16x16 a, bf16x16 b, floatx8 c) {
  return __builtin_amdgcn_wmma_f32_16x16x32_bf16(false, a, false, b, (short)0, c, false, false);
}

__global__ void cvt_f32_bf16(const float* __restrict__ in, bf16* __restrict__ out, int n) {
  int i = blockIdx.x * 256 + threadIdx.x;
  if (i < n) out[i] = (bf16)in[i];
}

// D = X * W^T + bias, X:[8192,1024] bf16, W:[1024,1024] bf16 (row-major).
// mode 0/1: write bf16 [B,H,T,dk]   (Q / K)
// mode 2  : write bf16 [B,H,dk,T]   (V transposed for attention B-fragments)
// mode 3  : write fp32 [B*T,F] with +residual (O-projection, pre-LN)
__global__ __launch_bounds__(256) void gemm_xwT(
    const bf16* __restrict__ X, const bf16* __restrict__ W,
    const float* __restrict__ bias, void* __restrict__ outp,
    const float* __restrict__ residual, int mode)
{
  const int lane  = threadIdx.x & 31;
  const int wid   = threadIdx.x >> 5;
  const int waveM = wid & 3, waveN = wid >> 2;
  const int row0  = blockIdx.x * 128 + waveM * 32;
  const int col0  = blockIdx.y * 64  + waveN * 32;

  floatx8 acc[2][2];
  acc[0][0] = zero8(); acc[0][1] = zero8();
  acc[1][0] = zero8(); acc[1][1] = zero8();

  for (int k0 = 0; k0 < F_DIM; k0 += 32) {
    bf16x16 a0 = load_frag(X, F_DIM, row0,      k0, lane);
    bf16x16 a1 = load_frag(X, F_DIM, row0 + 16, k0, lane);
    bf16x16 b0 = load_frag(W, F_DIM, col0,      k0, lane);
    bf16x16 b1 = load_frag(W, F_DIM, col0 + 16, k0, lane);
    acc[0][0] = wmma_bf16(a0, b0, acc[0][0]);
    acc[0][1] = wmma_bf16(a0, b1, acc[0][1]);
    acc[1][0] = wmma_bf16(a1, b0, acc[1][0]);
    acc[1][1] = wmma_bf16(a1, b1, acc[1][1]);
  }

  const int half = lane >> 4;
  const int nlo  = lane & 15;
  for (int i = 0; i < 2; ++i)
    for (int j = 0; j < 2; ++j)
      for (int v = 0; v < 8; ++v) {
        int m = row0 + i * 16 + v + 8 * half;       // C-layout row
        int n = col0 + j * 16 + nlo;                // C-layout col
        float val = acc[i][j][v] + bias[n];
        if (mode == 3) {
          float* O = (float*)outp;
          size_t idx = (size_t)m * F_DIM + n;
          O[idx] = val + residual[idx];
        } else {
          bf16* O = (bf16*)outp;
          int b = m >> 11, t = m & (T_DIM - 1);
          int h = n >> 6,  d = n & (DK_DIM - 1);
          size_t idx;
          if (mode == 2) idx = (((size_t)b * H_DIM + h) * DK_DIM + d) * T_DIM + t;
          else           idx = (((size_t)b * H_DIM + h) * T_DIM + t) * DK_DIM + d;
          O[idx] = (bf16)val;
        }
      }
}

// Flash attention: one wave handles a 16-query tile for one (b,h),
// streaming keys in chunks of 32 with online softmax (fp32 stats).
__global__ __launch_bounds__(128) void attn_fwd(
    const bf16* __restrict__ Qb, const bf16* __restrict__ Kb,
    const bf16* __restrict__ Vtb, bf16* __restrict__ Ctx)
{
  __shared__ __align__(16) bf16 Plds[4][16 * 32];
  const int lane  = threadIdx.x & 31;
  const int wid   = threadIdx.x >> 5;
  const int bh    = blockIdx.y;
  const int b     = bh >> 4, h = bh & 15;
  const int qrow0 = (blockIdx.x * 4 + wid) * 16;

  const bf16* Q  = Qb  + (size_t)bh * T_DIM * DK_DIM;
  const bf16* K  = Kb  + (size_t)bh * T_DIM * DK_DIM;
  const bf16* Vt = Vtb + (size_t)bh * DK_DIM * T_DIM;

  bf16x16 qf0 = load_frag(Q, DK_DIM, qrow0, 0,  lane);
  bf16x16 qf1 = load_frag(Q, DK_DIM, qrow0, 32, lane);

  float mrow[8], lrow[8];
  floatx8 octx[4];
  for (int v = 0; v < 8; ++v) { mrow[v] = -3.0e38f; lrow[v] = 0.f; }
  for (int j = 0; j < 4; ++j) octx[j] = zero8();

  const int half = lane >> 4;
  const int nlo  = lane & 15;
  const float invscale = 0.125f;  // 1/sqrt(dk)
  bf16* P = Plds[wid];

  for (int kt = 0; kt < T_DIM; kt += 32) {
    floatx8 s0 = zero8(), s1 = zero8();
    s0 = wmma_bf16(qf0, load_frag(K, DK_DIM, kt,      0,  lane), s0);
    s0 = wmma_bf16(qf1, load_frag(K, DK_DIM, kt,      32, lane), s0);
    s1 = wmma_bf16(qf0, load_frag(K, DK_DIM, kt + 16, 0,  lane), s1);
    s1 = wmma_bf16(qf1, load_frag(K, DK_DIM, kt + 16, 32, lane), s1);

    float p0[8], p1[8];
    for (int v = 0; v < 8; ++v) {
      float a = s0[v] * invscale;
      float c = s1[v] * invscale;
      float cm = fmaxf(a, c);
      for (int off = 1; off < 16; off <<= 1)       // 16-lane row reduce (wave32)
        cm = fmaxf(cm, __shfl_xor(cm, off, 32));
      float nm    = fmaxf(mrow[v], cm);
      float alpha = __expf(mrow[v] - nm);
      float e0 = __expf(a - nm);
      float e1 = __expf(c - nm);
      float rs = e0 + e1;
      for (int off = 1; off < 16; off <<= 1)
        rs += __shfl_xor(rs, off, 32);
      lrow[v] = lrow[v] * alpha + rs;
      mrow[v] = nm;
      for (int j = 0; j < 4; ++j) octx[j][v] *= alpha;
      p0[v] = e0; p1[v] = e1;
    }

    // Stage P (C-layout regs) -> LDS -> reload as A-fragment for P*V
    for (int v = 0; v < 8; ++v) {
      int r = v + 8 * half;
      P[r * 32 + nlo]      = (bf16)p0[v];
      P[r * 32 + 16 + nlo] = (bf16)p1[v];
    }
    asm volatile("s_wait_dscnt 0" ::: "memory");   // CDNA5 split DS counter
    FragU pf;
    pf.q[0] = *(const uintx4*)&P[(lane & 15) * 32 +      ((lane >> 4) << 3)];
    pf.q[1] = *(const uintx4*)&P[(lane & 15) * 32 + 16 + ((lane >> 4) << 3)];

    for (int j = 0; j < 4; ++j) {
      bf16x16 vf = load_frag(Vt, T_DIM, j * 16, kt, lane);  // rows = dk cols of V
      octx[j] = wmma_bf16(pf.v, vf, octx[j]);
    }
  }

  float rl[8];
  for (int v = 0; v < 8; ++v) rl[v] = 1.0f / lrow[v];
  for (int j = 0; j < 4; ++j)
    for (int v = 0; v < 8; ++v) {
      int t = qrow0 + v + 8 * half;
      int f = h * DK_DIM + j * 16 + nlo;
      Ctx[((size_t)b * T_DIM + t) * F_DIM + f] = (bf16)(octx[j][v] * rl[v]);
    }
}

__global__ __launch_bounds__(256) void layernorm_res(
    const float* __restrict__ x, const float* __restrict__ g,
    const float* __restrict__ beta, float* __restrict__ out)
{
  __shared__ float ssum[256], ssq[256];
  const int row = blockIdx.x;
  const float* xr = x + (size_t)row * F_DIM;
  float vv[4]; float s = 0.f, s2 = 0.f;
  for (int i = 0; i < 4; ++i) {
    vv[i] = xr[threadIdx.x + i * 256];
    s += vv[i]; s2 += vv[i] * vv[i];
  }
  ssum[threadIdx.x] = s; ssq[threadIdx.x] = s2;
  __syncthreads();
  for (int o = 128; o > 0; o >>= 1) {
    if ((int)threadIdx.x < o) {
      ssum[threadIdx.x] += ssum[threadIdx.x + o];
      ssq[threadIdx.x]  += ssq[threadIdx.x + o];
    }
    __syncthreads();
  }
  float mu   = ssum[0] * (1.0f / F_DIM);
  float var  = ssq[0]  * (1.0f / F_DIM) - mu * mu;
  float rstd = rsqrtf(var + 1e-5f);
  for (int i = 0; i < 4; ++i) {
    int c = threadIdx.x + i * 256;
    out[(size_t)row * F_DIM + c] = (vv[i] - mu) * rstd * g[c] + beta[c];
  }
}

extern "C" void kernel_launch(void* const* d_in, const int* in_sizes, int n_in,
                              void* d_out, int out_size, void* d_ws, size_t ws_size,
                              hipStream_t stream) {
  const float* key   = (const float*)d_in[0];
  const float* value = (const float*)d_in[1];
  const float* query = (const float*)d_in[2];
  const float* Wq = (const float*)d_in[3];
  const float* bq = (const float*)d_in[4];
  const float* Wk = (const float*)d_in[5];
  const float* bk = (const float*)d_in[6];
  const float* Wv = (const float*)d_in[7];
  const float* bv = (const float*)d_in[8];
  const float* Wo = (const float*)d_in[9];
  const float* bo = (const float*)d_in[10];
  const float* ln_g = (const float*)d_in[11];
  const float* ln_b = (const float*)d_in[12];

  char* ws = (char*)d_ws;
  const size_t MB = 1ull << 20;
  bf16* qb  = (bf16*)(ws + 0);         // [B,H,T,dk]  16 MB
  bf16* kb  = (bf16*)(ws + 16 * MB);   // [B,H,T,dk]  16 MB
  bf16* vtb = (bf16*)(ws + 32 * MB);   // [B,H,dk,T]  16 MB
  bf16* qx  = (bf16*)(ws + 48 * MB);   // query bf16  16 MB
  bf16* kx  = (bf16*)(ws + 64 * MB);   // key   bf16  16 MB
  bf16* vx  = (bf16*)(ws + 80 * MB);   // value bf16  16 MB
  bf16* wqb = (bf16*)(ws + 96 * MB);
  bf16* wkb = (bf16*)(ws + 98 * MB);
  bf16* wvb = (bf16*)(ws + 100 * MB);
  bf16* wob = (bf16*)(ws + 102 * MB);  // peak usage = 104 MB
  bf16*  ctx  = (bf16*)(ws + 48 * MB); // reuse qx after Q-projection done
  float* xres = (float*)(ws + 64 * MB);// reuse kx/vx after projections done

  const int NA = B_DIM * T_DIM * F_DIM;  // 8388608 activation elements
  const int NW = F_DIM * F_DIM;          // 1048576 weight elements
  cvt_f32_bf16<<<(NA + 255) / 256, 256, 0, stream>>>(query, qx, NA);
  cvt_f32_bf16<<<(NA + 255) / 256, 256, 0, stream>>>(key,   kx, NA);
  cvt_f32_bf16<<<(NA + 255) / 256, 256, 0, stream>>>(value, vx, NA);
  cvt_f32_bf16<<<(NW + 255) / 256, 256, 0, stream>>>(Wq, wqb, NW);
  cvt_f32_bf16<<<(NW + 255) / 256, 256, 0, stream>>>(Wk, wkb, NW);
  cvt_f32_bf16<<<(NW + 255) / 256, 256, 0, stream>>>(Wv, wvb, NW);
  cvt_f32_bf16<<<(NW + 255) / 256, 256, 0, stream>>>(Wo, wob, NW);

  dim3 ggrid(64, 16), gblk(256);
  gemm_xwT<<<ggrid, gblk, 0, stream>>>(qx, wqb, bq, qb,  nullptr, 0);
  gemm_xwT<<<ggrid, gblk, 0, stream>>>(kx, wkb, bk, kb,  nullptr, 1);
  gemm_xwT<<<ggrid, gblk, 0, stream>>>(vx, wvb, bv, vtb, nullptr, 2);

  dim3 agrid(T_DIM / 64, B_DIM * H_DIM), ablk(128);
  attn_fwd<<<agrid, ablk, 0, stream>>>(qb, kb, vtb, ctx);

  gemm_xwT<<<ggrid, gblk, 0, stream>>>(ctx, wob, bo, xres, query, 3);

  layernorm_res<<<B_DIM * T_DIM, 256, 0, stream>>>(xres, ln_g, ln_b, (float*)d_out);
}
